// LSTMModel_13245679141150
// MI455X (gfx1250) — compile-verified, matrix operand order checked by reading, not run on hardware
//
#include <hip/hip_runtime.h>
#include <hip/hip_bf16.h>

#define B_   256
#define T_   256
#define D_   64
#define H_   512
#define G_   2048   // 4*H

typedef __bf16 bf16;
typedef bf16  v16bf __attribute__((ext_vector_type(16)));
typedef bf16  v8bf  __attribute__((ext_vector_type(8)));
typedef float v8f   __attribute__((ext_vector_type(8)));

static __device__ __forceinline__ v16bf join16(v8bf lo, v8bf hi) {
    return __builtin_shufflevector(lo, hi, 0,1,2,3,4,5,6,7,8,9,10,11,12,13,14,15);
}
static __device__ __forceinline__ float sigmoidf(float x) {
    return 1.0f / (1.0f + __expf(-x));
}
#if __has_builtin(__builtin_amdgcn_tanhf)
static __device__ __forceinline__ float fast_tanh(float x) { return __builtin_amdgcn_tanhf(x); }
#else
static __device__ __forceinline__ float fast_tanh(float x) { return tanhf(x); }
#endif

// Soft scheduler hint: per k-step form {2 DS reads, 8 VMEM reads, 4 WMMA}.
// This yields clause-batched loads with staggered waits (loadcnt 6/4/2/0)
// across the WMMA group, with no extra register pressure (no spills).
static __device__ __forceinline__ void sched_pipe() {
    __builtin_amdgcn_sched_group_barrier(0x100, 2, 0); // DS read  (A fragments)
    __builtin_amdgcn_sched_group_barrier(0x020, 8, 0); // VMEM read (B fragments)
    __builtin_amdgcn_sched_group_barrier(0x008, 4, 0); // MFMA/WMMA
}

static __device__ __forceinline__ v16bf loadA(const bf16* __restrict__ arow, int k) {
    return join16(*(const v8bf*)(arow + k), *(const v8bf*)(arow + k + 16));
}
static __device__ __forceinline__ void loadB4(v16bf* __restrict__ b,
                                              const bf16* const* __restrict__ wrow, int k) {
#pragma unroll
    for (int g = 0; g < 4; ++g) {
        const bf16* wp = wrow[g] + k;
        b[g] = join16(*(const v8bf*)wp, *(const v8bf*)(wp + 16));
    }
}

// ---------------- preprocessing kernels ----------------

__global__ void cvt_f32_bf16(const float* __restrict__ src, bf16* __restrict__ dst, int n) {
    int i = blockIdx.x * blockDim.x + threadIdx.x;
    int stride = gridDim.x * blockDim.x;
    for (; i < n; i += stride) dst[i] = (bf16)src[i];
}

template <int DIN>
__global__ void pack_weights(const float* __restrict__ Wih, const float* __restrict__ Whh,
                             bf16* __restrict__ dst) {
    const int KP = DIN + H_;
    int i = blockIdx.x * blockDim.x + threadIdx.x;
    int stride = gridDim.x * blockDim.x;
    int total = G_ * KP;
    for (; i < total; i += stride) {
        int n = i / KP, k = i - n * KP;
        float v = (k < DIN) ? Wih[n * DIN + k] : Whh[n * H_ + (k - DIN)];
        dst[i] = (bf16)v;
    }
}

__global__ void add_bias(const float* __restrict__ a, const float* __restrict__ b,
                         float* __restrict__ dst, int n) {
    int i = blockIdx.x * blockDim.x + threadIdx.x;
    if (i < n) dst[i] = a[i] + b[i];
}

// ---------------- software-pipelined 16xKLENx(4x16) GEMM panel ----------------
// FIRST: seed the k=0 WMMA group with the inline-0 C operand (no acc init movs).
template <int KLEN, bool FIRST>
static __device__ __forceinline__ void gemm_panel(const bf16* __restrict__ arow,
                                                  const bf16* const* __restrict__ wrow,
                                                  int wk0, v8f* __restrict__ acc) {
    v16bf a_cur = loadA(arow, 0);
    v16bf b_cur[4];
    loadB4(b_cur, wrow, wk0);
    v16bf a_nxt = loadA(arow, 32);          // over-read on tiny panels is dead/padded
    v16bf b_nxt[4];
    loadB4(b_nxt, wrow, wk0 + 32);
    if constexpr (FIRST) {
        v8f zc = {};
#pragma unroll
        for (int g = 0; g < 4; ++g)
            acc[g] = __builtin_amdgcn_wmma_f32_16x16x32_bf16(
                false, a_cur, false, b_cur[g], (short)0, zc, false, false);
    } else {
#pragma unroll
        for (int g = 0; g < 4; ++g)
            acc[g] = __builtin_amdgcn_wmma_f32_16x16x32_bf16(
                false, a_cur, false, b_cur[g], (short)0, acc[g], false, false);
    }
    sched_pipe();
#pragma unroll 2
    for (int k = 32; k < KLEN; k += 32) {
        a_cur = a_nxt;
#pragma unroll
        for (int g = 0; g < 4; ++g) b_cur[g] = b_nxt[g];
        a_nxt = loadA(arow, k + 32);        // final-iteration prefetch is dead (DCE/padding)
        loadB4(b_nxt, wrow, wk0 + k + 32);
#pragma unroll
        for (int g = 0; g < 4; ++g)
            acc[g] = __builtin_amdgcn_wmma_f32_16x16x32_bf16(
                false, a_cur, false, b_cur[g], (short)0, acc[g], false, false);
        sched_pipe();
    }
}

// ---------------- fused LSTM layer scan ----------------
// grid = 16 (batch blocks of 16 rows), block = 1024 (32 waves)
// wave w owns h columns [16w, 16w+16); computes gate tiles n = g*512 + 16w.
template <int DIN>
__launch_bounds__(1024)
__global__ void lstm_scan(const bf16*  __restrict__ xin,   // [B][T][DIN] bf16
                          const bf16*  __restrict__ Wp,    // [G][DIN+H] bf16 packed
                          const float* __restrict__ bias,  // [G] = b_ih + b_hh
                          bf16*        __restrict__ hout,  // [B][T][H] bf16 or null
                          float*       __restrict__ hlast) // [B][H] f32 or null
{
    constexpr int KP     = DIN + H_;
    constexpr int XPAD   = DIN + 8;        // LDS x-panel row stride (bf16)
    constexpr int HPAD   = H_ + 8;         // LDS h-panel row stride (bf16)
    constexpr int NCHUNK = 16 * DIN / 8;   // 16B async chunks per x panel

    __shared__ bf16 xs[2][16][XPAD];       // double-buffered x_t panel
    __shared__ bf16 hs[16][HPAD];          // h_{t-1} panel
    __shared__ bf16 pad_guard[64];         // soak up tail over-reads
    (void)pad_guard;

    const int tid  = threadIdx.x;
    const int lane = tid & 31;
    const int w    = tid >> 5;
    const int bm0  = blockIdx.x * 16;
    const int j0   = w * 16;
    const int mrow = lane & 15;            // A row / B col within tile
    const int koff = (lane >> 4) << 3;     // 0 or 8: K sub-chunk per ISA layout

    const bf16* wrow[4];
    float biasv[4];
#pragma unroll
    for (int g = 0; g < 4; ++g) {
        int n = g * H_ + j0 + (lane & 15);
        wrow[g]  = Wp + (size_t)n * KP + koff;
        biasv[g] = bias[n];
    }

    // h_{-1} = 0
    for (int i = tid; i < 16 * H_; i += blockDim.x) {
        int r = i >> 9, c = i & (H_ - 1);
        hs[r][c] = (bf16)0.0f;
    }
    // async prefetch x_0 into xs[0] (ASYNCcnt-tracked)
    if (tid < NCHUNK) {
        int r = tid / (DIN / 8);
        int c = (tid % (DIN / 8)) * 8;
        unsigned lds = (unsigned)(size_t)&xs[0][r][c];
        unsigned long long ga =
            (unsigned long long)(size_t)(xin + ((size_t)(bm0 + r) * T_) * DIN + c);
        asm volatile("global_load_async_to_lds_b128 %0, %1, off"
                     :: "v"(lds), "v"(ga) : "memory");
    }
    asm volatile("s_wait_asynccnt 0" ::: "memory");
    __syncthreads();

    v8f cprev;
#pragma unroll
    for (int e = 0; e < 8; ++e) cprev[e] = 0.0f;

    const int rbase = (lane >> 4) << 3;
    const int col   = lane & 15;

    for (int t = 0; t < T_; ++t) {
        const int buf = t & 1, nxt = buf ^ 1;

        // prefetch x_{t+1}, overlapped with this step's WMMA work
        if (t + 1 < T_ && tid < NCHUNK) {
            int r = tid / (DIN / 8);
            int c = (tid % (DIN / 8)) * 8;
            unsigned lds = (unsigned)(size_t)&xs[nxt][r][c];
            unsigned long long ga =
                (unsigned long long)(size_t)(xin + ((size_t)(bm0 + r) * T_ + (t + 1)) * DIN + c);
            asm volatile("global_load_async_to_lds_b128 %0, %1, off"
                         :: "v"(lds), "v"(ga) : "memory");
        }

        // gates = [x_t | h_{t-1}] . W^T  (bias folded into activation tail)
        v8f acc[4];
        gemm_panel<DIN, true>(&xs[buf][mrow][koff], wrow, 0, acc);
        gemm_panel<H_, false>(&hs[mrow][koff], wrow, DIN, acc);

        // elementwise cell update (C/D layout: row = e + 8*(lane>=16), col = lane&15)
        v8f hnew;
#pragma unroll
        for (int e = 0; e < 8; ++e) {
            float iv = sigmoidf(acc[0][e] + biasv[0]);
            float fv = sigmoidf(acc[1][e] + biasv[1]);
            float gv = fast_tanh(acc[2][e] + biasv[2]);
            float ov = sigmoidf(acc[3][e] + biasv[3]);
            float c  = fv * cprev[e] + iv * gv;
            cprev[e] = c;
            hnew[e]  = ov * fast_tanh(c);
        }

        __syncthreads();   // all reads of hs done before overwriting with h_t

#pragma unroll
        for (int e = 0; e < 8; ++e)
            hs[e + rbase][j0 + col] = (bf16)hnew[e];
        if (hout) {
#pragma unroll
            for (int e = 0; e < 8; ++e)
                hout[((size_t)(bm0 + e + rbase) * T_ + t) * H_ + j0 + col] = (bf16)hnew[e];
        }
        if (hlast && t == T_ - 1) {
#pragma unroll
            for (int e = 0; e < 8; ++e)
                hlast[(size_t)(bm0 + e + rbase) * H_ + j0 + col] = hnew[e];
        }

        asm volatile("s_wait_asynccnt 0" ::: "memory");  // x_{t+1} landed in LDS
        __syncthreads();
    }
}

// ---------------- final FC (H=512 -> 1), one wave per batch row ----------------
__global__ void fc_kernel(const float* __restrict__ h, const float* __restrict__ wfc,
                          const float* __restrict__ bfc, float* __restrict__ out) {
    int w = threadIdx.x >> 5, lane = threadIdx.x & 31;
    int b = blockIdx.x * 8 + w;
    float s = 0.0f;
    for (int j = lane; j < H_; j += 32) s += h[(size_t)b * H_ + j] * wfc[j];
#pragma unroll
    for (int off = 16; off > 0; off >>= 1) s += __shfl_down(s, off, 32);
    if (lane == 0) out[b] = s + bfc[0];
}

// ---------------- launcher ----------------
extern "C" void kernel_launch(void* const* d_in, const int* in_sizes, int n_in,
                              void* d_out, int out_size, void* d_ws, size_t ws_size,
                              hipStream_t stream) {
    (void)in_sizes; (void)n_in; (void)out_size; (void)ws_size;
    const float* x    = (const float*)d_in[0];
    const float* Wih0 = (const float*)d_in[1];
    const float* Whh0 = (const float*)d_in[2];
    const float* bih0 = (const float*)d_in[3];
    const float* bhh0 = (const float*)d_in[4];
    const float* Wih1 = (const float*)d_in[5];
    const float* Whh1 = (const float*)d_in[6];
    const float* bih1 = (const float*)d_in[7];
    const float* bhh1 = (const float*)d_in[8];
    const float* Wfc  = (const float*)d_in[9];
    const float* bfc  = (const float*)d_in[10];
    float* out = (float*)d_out;

    char* ws = (char*)d_ws;
    size_t off = 0;
    auto alloc = [&](size_t bytes) -> void* {
        void* p = ws + off;
        off = (off + bytes + 255) & ~(size_t)255;
        return p;
    };
    bf16*  xb    = (bf16*)alloc((size_t)B_ * T_ * D_ * 2);        // 8 MB
    bf16*  W0    = (bf16*)alloc((size_t)G_ * (D_ + H_) * 2);      // 2.25 MB
    bf16*  W1    = (bf16*)alloc((size_t)G_ * (H_ + H_) * 2);      // 4 MB
    float* bias0 = (float*)alloc((size_t)G_ * 4);
    float* bias1 = (float*)alloc((size_t)G_ * 4);
    bf16*  h1    = (bf16*)alloc((size_t)B_ * T_ * H_ * 2);        // 64 MB
    float* h2    = (float*)alloc((size_t)B_ * H_ * 4);            // 0.5 MB

    cvt_f32_bf16<<<256, 256, 0, stream>>>(x, xb, B_ * T_ * D_);
    pack_weights<D_><<<512, 256, 0, stream>>>(Wih0, Whh0, W0);
    pack_weights<H_><<<512, 256, 0, stream>>>(Wih1, Whh1, W1);
    add_bias<<<8, 256, 0, stream>>>(bih0, bhh0, bias0, G_);
    add_bias<<<8, 256, 0, stream>>>(bih1, bhh1, bias1, G_);

    lstm_scan<D_><<<16, 1024, 0, stream>>>(xb, W0, bias0, h1, nullptr);
    lstm_scan<H_><<<16, 1024, 0, stream>>>(h1, W1, bias1, nullptr, h2);

    fc_kernel<<<32, 256, 0, stream>>>(h2, Wfc, bfc, out);
}
